// Aattn_72344429133937
// MI455X (gfx1250) — matrix operand order repaired and checked by reading, not compiled, not run on hardware
//
#include <hip/hip_runtime.h>
#include <hip/hip_bf16.h>
#include <math.h>

typedef __attribute__((ext_vector_type(16))) _Float16 v16h;
typedef __attribute__((ext_vector_type(8)))  float    v8f;

#define C_DIM 512
#define N_DIM 4096
#define B_DIM 2

// ---------------------------------------------------------------------------
// WMMA wrapper: D = A(16x32 f16) * B(32x16 f16) + C(16x16 f32)
// ---------------------------------------------------------------------------
__device__ __forceinline__ v8f wmma_f16(v16h a, v16h b, v8f c) {
    return __builtin_amdgcn_wmma_f32_16x16x32_f16(
        /*neg_a=*/false, a, /*neg_b=*/false, b,
        /*c_mod=*/(short)0, c, /*reuse_a=*/false, /*reuse_b=*/false);
}

// A-matrix 16x32 f16 operand, contiguous k order used by this file (B matches):
//   row M = lane%16 ; lanes<16: K = {0..15}; lanes>=16: K = {16..31}
// element (row,k) lives at base[row*row_stride + k]
__device__ __forceinline__ v16h load_a_f16(const _Float16* base, int row_stride, int lane) {
    const _Float16* rp = base + (size_t)(lane & 15) * row_stride;
    const int ko = (lane >= 16) ? 16 : 0;
    v16h a;
#pragma unroll
    for (int j = 0; j < 16; ++j) a[j] = rp[ko + j];
    return a;
}

// B-matrix 32x16 f16 layout: col N = lane%16; lanes<16: K=j; lanes>=16: K=16+j.
// "k-contiguous": element (k,n) at base[n*col_stride + k]
__device__ __forceinline__ v16h load_b_kcontig_f16(const _Float16* base, int col_stride, int lane) {
    const _Float16* p = base + (size_t)(lane & 15) * col_stride + ((lane >= 16) ? 16 : 0);
    v16h b;
#pragma unroll
    for (int j = 0; j < 16; ++j) b[j] = p[j];
    return b;
}

// ---------------------------------------------------------------------------
// Kernel 0: f32 -> f16 conversion (weights), done once instead of per-block.
// ---------------------------------------------------------------------------
__global__ void __launch_bounds__(256)
wcvt_kernel(const float* __restrict__ src, _Float16* __restrict__ dst, int n) {
    const int i = blockIdx.x * 256 + threadIdx.x;
    if (i < n) dst[i] = (_Float16)src[i];
}

// ---------------------------------------------------------------------------
// Kernel 1: 1x1 conv projection, f32 in (channel-major) -> f16 out.
//   out[n][o] = sum_c w[o][c] * in[b][c][n] + bias[o]
// A = f16 weights (rows o, k-contiguous), B = input staged through LDS
// (cooperative coalesced f32 load, converted once, row stride 40 halves =>
// 16B-aligned, conflict-free). D rows = o, cols = n.
// transpose_out: 0 -> out[(b,n),(o)] stride C ; 1 -> out[(b,o),(n)] stride N
// ---------------------------------------------------------------------------
__global__ void __launch_bounds__(256)
proj_kernel(const float* __restrict__ in, const _Float16* __restrict__ wf,
            const float* __restrict__ bias, _Float16* __restrict__ out,
            int transpose_out) {
    __shared__ _Float16 Bs[16 * 40];  // [n][c], c-padded to 40

    const int blk   = blockIdx.x;
    const int b     = blk / (N_DIM / 16);
    const int ntile = (blk % (N_DIM / 16)) * 16;
    const int lane  = threadIdx.x & 31;
    const int wave  = threadIdx.x >> 5;
    const int obase = wave * 64;

    const float* inb = in + (size_t)b * C_DIM * N_DIM;
    v8f acc[4] = {};

    for (int cs = 0; cs < C_DIM; cs += 32) {
        __syncthreads();
        // stage 32(c) x 16(n) f32 tile -> f16 LDS, coalesced on n
        for (int idx = threadIdx.x; idx < 512; idx += 256) {
            const int n = idx & 15, c = idx >> 4;
            Bs[n * 40 + c] = (_Float16)inb[(size_t)(cs + c) * N_DIM + ntile + n];
        }
        __syncthreads();
        v16h bt = load_b_kcontig_f16(Bs, 40, lane);
#pragma unroll
        for (int ct = 0; ct < 4; ++ct) {
            v16h at = load_a_f16(wf + (size_t)(obase + ct * 16) * C_DIM + cs, C_DIM, lane);
            acc[ct] = wmma_f16(at, bt, acc[ct]);
        }
    }

    const int n     = ntile + (lane & 15);
    const int rbase = (lane >= 16) ? 8 : 0;
#pragma unroll
    for (int ct = 0; ct < 4; ++ct) {
#pragma unroll
        for (int r = 0; r < 8; ++r) {
            const int o = obase + ct * 16 + rbase + r;
            const float v = acc[ct][r] + bias[o];
            if (transpose_out)
                out[((size_t)b * C_DIM + o) * N_DIM + n] = (_Float16)v;
            else
                out[((size_t)b * N_DIM + n) * C_DIM + o] = (_Float16)v;
        }
    }
}

// ---------------------------------------------------------------------------
// Kernel 2: per-(b,c) mean and 1/sqrt(var_unbiased + eps) of content spatially
// ---------------------------------------------------------------------------
__global__ void __launch_bounds__(256)
stats_kernel(const float* __restrict__ content,
             float* __restrict__ cmean, float* __restrict__ cinvstd) {
    const int bc = blockIdx.x;  // b*C + c
    const float* p = content + (size_t)bc * N_DIM;
    float s = 0.f, sq = 0.f;
    for (int i = threadIdx.x; i < N_DIM; i += 256) {
        float x = p[i]; s += x; sq += x * x;
    }
    __shared__ float rs[256], rq[256];
    rs[threadIdx.x] = s; rq[threadIdx.x] = sq;
    __syncthreads();
    for (int st = 128; st > 0; st >>= 1) {
        if ((int)threadIdx.x < st) {
            rs[threadIdx.x] += rs[threadIdx.x + st];
            rq[threadIdx.x] += rq[threadIdx.x + st];
        }
        __syncthreads();
    }
    if (threadIdx.x == 0) {
        float mean = rs[0] / (float)N_DIM;
        float var  = (rq[0] - (float)N_DIM * mean * mean) / (float)(N_DIM - 1);
        cmean[bc]   = mean;
        cinvstd[bc] = rsqrtf(var + 1e-5f);
    }
}

// ---------------------------------------------------------------------------
// Kernel 3: softmax statistics (row max, row sum of exp) per query row.
// Block = 8 waves, one 16-query tile. Q tile (16x512 f16 = 16KB) staged in LDS
// once and shared by all waves. Wave w streams keys w*32 + it*256. Online
// (m,s) per row in registers, 16-lane shfl_xor reductions, LDS merge.
// ---------------------------------------------------------------------------
__global__ void __launch_bounds__(256)
smstats_kernel(const _Float16* __restrict__ Q, const _Float16* __restrict__ K,
               float* __restrict__ rowmax, float* __restrict__ rowsum) {
    __shared__ _Float16 Qs[16 * C_DIM];
    __shared__ float lm[8][16], ls[8][16];

    const int blk   = blockIdx.x;
    const int b     = blk / (N_DIM / 16);
    const int qtile = (blk % (N_DIM / 16)) * 16;
    const int lane  = threadIdx.x & 31;
    const int wave  = threadIdx.x >> 5;

    // stage Q tile (coalesced dword copy)
    {
        const unsigned* src = (const unsigned*)(Q + ((size_t)b * N_DIM + qtile) * C_DIM);
        unsigned* dst = (unsigned*)Qs;
        for (int i = threadIdx.x; i < 16 * C_DIM / 2; i += 256) dst[i] = src[i];
    }
    __syncthreads();

    const _Float16* Kb = K + (size_t)b * N_DIM * C_DIM;

    float m[8], s[8];
#pragma unroll
    for (int r = 0; r < 8; ++r) { m[r] = -1e30f; s[r] = 0.f; }

    for (int it = 0; it < N_DIM / 256; ++it) {
        const int kb = it * 256 + wave * 32;
#pragma unroll
        for (int half = 0; half < 2; ++half) {
            const int k0 = kb + half * 16;
            v8f d = {};
            for (int cs = 0; cs < C_DIM; cs += 32) {
                v16h a  = load_a_f16(Qs + cs, C_DIM, lane);  // LDS
                v16h bt = load_b_kcontig_f16(Kb + (size_t)k0 * C_DIM + cs, C_DIM, lane);
                d = wmma_f16(a, bt, d);
            }
#pragma unroll
            for (int r = 0; r < 8; ++r) {
                float v = d[r];
                float tm = v;
                tm = fmaxf(tm, __shfl_xor(tm, 1, 32));
                tm = fmaxf(tm, __shfl_xor(tm, 2, 32));
                tm = fmaxf(tm, __shfl_xor(tm, 4, 32));
                tm = fmaxf(tm, __shfl_xor(tm, 8, 32));
                const float nm = fmaxf(m[r], tm);
                float p = __expf(v - nm);
                p += __shfl_xor(p, 1, 32);
                p += __shfl_xor(p, 2, 32);
                p += __shfl_xor(p, 4, 32);
                p += __shfl_xor(p, 8, 32);
                s[r] = s[r] * __expf(m[r] - nm) + p;
                m[r] = nm;
            }
        }
    }

    if (lane == 0 || lane == 16) {
        const int rbase = (lane == 16) ? 8 : 0;
#pragma unroll
        for (int r = 0; r < 8; ++r) { lm[wave][rbase + r] = m[r]; ls[wave][rbase + r] = s[r]; }
    }
    __syncthreads();
    if (threadIdx.x < 16) {
        const int row = threadIdx.x;
        float M = -1e30f;
        for (int w2 = 0; w2 < 8; ++w2) M = fmaxf(M, lm[w2][row]);
        float S = 0.f;
        for (int w2 = 0; w2 < 8; ++w2) S += ls[w2][row] * __expf(lm[w2][row] - M);
        rowmax[(size_t)b * N_DIM + qtile + row] = M;
        rowsum[(size_t)b * N_DIM + qtile + row] = S;
    }
}

// ---------------------------------------------------------------------------
// Kernel 4: P*V and P*V^2 accumulation. Q tile staged in LDS (16KB) and shared
// by all 8 waves; per iteration the waves cooperatively build a 16x256 P tile
// in LDS (each wave 32 keys); each wave then accumulates its own 64-channel
// slice of E[V] and E[V^2]. Epilogue fuses std/mean + mean-variance-norm of
// content -> T (f16, [b][n][c]).
// ---------------------------------------------------------------------------
__global__ void __launch_bounds__(256)
pv_kernel(const _Float16* __restrict__ Q, const _Float16* __restrict__ K,
          const _Float16* __restrict__ VT,
          const float* __restrict__ rowmax, const float* __restrict__ rowsum,
          const float* __restrict__ content,
          const float* __restrict__ cmean, const float* __restrict__ cinvstd,
          _Float16* __restrict__ T) {
    __shared__ _Float16 Qs[16 * C_DIM];  // 16 KB
    __shared__ _Float16 P[16 * 256];     //  8 KB, row-major [qrow][key]

    const int blk   = blockIdx.x;
    const int b     = blk / (N_DIM / 16);
    const int qtile = (blk % (N_DIM / 16)) * 16;
    const int lane  = threadIdx.x & 31;
    const int wave  = threadIdx.x >> 5;
    const int rbase = (lane >= 16) ? 8 : 0;

    {
        const unsigned* src = (const unsigned*)(Q + ((size_t)b * N_DIM + qtile) * C_DIM);
        unsigned* dst = (unsigned*)Qs;
        for (int i = threadIdx.x; i < 16 * C_DIM / 2; i += 256) dst[i] = src[i];
    }
    __syncthreads();

    const _Float16* Kb = K  + (size_t)b * N_DIM * C_DIM;
    const _Float16* Vb = VT + (size_t)b * C_DIM * N_DIM;

    float Mv[8], Sinv[8];
#pragma unroll
    for (int r = 0; r < 8; ++r) {
        const int row = qtile + rbase + r;
        Mv[r]   = rowmax[(size_t)b * N_DIM + row];
        Sinv[r] = 1.0f / rowsum[(size_t)b * N_DIM + row];
    }

    const int cs0 = wave * 64;
    v8f accM[4] = {}, accS[4] = {};

    for (int it = 0; it < N_DIM / 256; ++it) {
        const int keybase = it * 256;
        // --- build this wave's 32 columns of P ---
#pragma unroll
        for (int half = 0; half < 2; ++half) {
            const int k0 = keybase + wave * 32 + half * 16;
            v8f d = {};
            for (int cs = 0; cs < C_DIM; cs += 32) {
                v16h a  = load_a_f16(Qs + cs, C_DIM, lane);  // LDS
                v16h bt = load_b_kcontig_f16(Kb + (size_t)k0 * C_DIM + cs, C_DIM, lane);
                d = wmma_f16(a, bt, d);
            }
            const int col = wave * 32 + half * 16 + (lane & 15);
#pragma unroll
            for (int r = 0; r < 8; ++r) {
                const float p = __expf(d[r] - Mv[r]) * Sinv[r];
                P[(rbase + r) * 256 + col] = (_Float16)p;
            }
        }
        __syncthreads();
        // --- consume the 16x256 P tile against this wave's channel slice ---
#pragma unroll
        for (int ks = 0; ks < 8; ++ks) {
            v16h pa = load_a_f16(P + ks * 32, 256, lane);  // LDS A-tile
#pragma unroll
            for (int ct = 0; ct < 4; ++ct) {
                const int c0 = cs0 + ct * 16;
                v16h vb = load_b_kcontig_f16(Vb + (size_t)c0 * N_DIM + keybase + ks * 32,
                                             N_DIM, lane);
                v16h vb2;
#pragma unroll
                for (int j = 0; j < 16; ++j) vb2[j] = vb[j] * vb[j];
                accM[ct] = wmma_f16(pa, vb,  accM[ct]);
                accS[ct] = wmma_f16(pa, vb2, accS[ct]);
            }
        }
        __syncthreads();
    }

    // Epilogue: D rows = query, cols = channel (lane%16).
    const float* cb = content + (size_t)b * C_DIM * N_DIM;
#pragma unroll
    for (int ct = 0; ct < 4; ++ct) {
        const int c = cs0 + ct * 16 + (lane & 15);
        const float cm = cmean[(size_t)b * C_DIM + c];
        const float ci = cinvstd[(size_t)b * C_DIM + c];
#pragma unroll
        for (int r = 0; r < 8; ++r) {
            const int n = qtile + rbase + r;
            const float mv  = accM[ct][r];
            const float sec = accS[ct][r];
            const float sd  = sqrtf(fmaxf(sec - mv * mv, 0.f));
            const float x   = cb[(size_t)c * N_DIM + n];
            const float t   = sd * (x - cm) * ci + mv;
            T[((size_t)b * N_DIM + n) * C_DIM + c] = (_Float16)t;
        }
    }
}

// ---------------------------------------------------------------------------
// Kernel 5: out conv (T[n][c] x out_w^T f16) + bias + residual -> f32 [b][o][n]
// T tile staged in LDS (16KB) and shared by all 8 waves.
// ---------------------------------------------------------------------------
__global__ void __launch_bounds__(256)
outconv_kernel(const _Float16* __restrict__ T, const _Float16* __restrict__ wf,
               const float* __restrict__ bias, const float* __restrict__ content,
               float* __restrict__ out) {
    __shared__ _Float16 Ts[16 * C_DIM];  // 16 KB

    const int blk   = blockIdx.x;
    const int b     = blk / (N_DIM / 16);
    const int ntile = (blk % (N_DIM / 16)) * 16;
    const int lane  = threadIdx.x & 31;
    const int wave  = threadIdx.x >> 5;
    const int obase = wave * 64;
    const int rbase = (lane >= 16) ? 8 : 0;

    {
        const unsigned* src = (const unsigned*)(T + ((size_t)b * N_DIM + ntile) * C_DIM);
        unsigned* dst = (unsigned*)Ts;
        for (int i = threadIdx.x; i < 16 * C_DIM / 2; i += 256) dst[i] = src[i];
    }
    __syncthreads();

    v8f acc[4] = {};
    for (int cs = 0; cs < C_DIM; cs += 32) {
        v16h a = load_a_f16(Ts + cs, C_DIM, lane);  // LDS; rows = n, k = c
#pragma unroll
        for (int ct = 0; ct < 4; ++ct) {
            v16h bt = load_b_kcontig_f16(wf + (size_t)(obase + ct * 16) * C_DIM + cs,
                                         C_DIM, lane);
            acc[ct] = wmma_f16(a, bt, acc[ct]);
        }
    }

#pragma unroll
    for (int ct = 0; ct < 4; ++ct) {
        const int o = obase + ct * 16 + (lane & 15);
        const float bo = bias[o];
#pragma unroll
        for (int r = 0; r < 8; ++r) {
            const int n = ntile + rbase + r;
            const size_t idx = ((size_t)b * C_DIM + o) * N_DIM + n;
            out[idx] = acc[ct][r] + bo + content[idx];
        }
    }
}

// ---------------------------------------------------------------------------
extern "C" void kernel_launch(void* const* d_in, const int* in_sizes, int n_in,
                              void* d_out, int out_size, void* d_ws, size_t ws_size,
                              hipStream_t stream) {
    const float* content = (const float*)d_in[0];
    const float* style   = (const float*)d_in[1];
    const float* f_w = (const float*)d_in[2];
    const float* f_b = (const float*)d_in[3];
    const float* g_w = (const float*)d_in[4];
    const float* g_b = (const float*)d_in[5];
    const float* h_w = (const float*)d_in[6];
    const float* h_b = (const float*)d_in[7];
    const float* o_w = (const float*)d_in[8];
    const float* o_b = (const float*)d_in[9];
    float* out = (float*)d_out;

    char* ws = (char*)d_ws;
    const size_t halfBuf = (size_t)B_DIM * N_DIM * C_DIM * sizeof(_Float16);  // 8 MB
    const size_t wElems  = (size_t)C_DIM * C_DIM;
    _Float16* Q   = (_Float16*)(ws);
    _Float16* Kb  = (_Float16*)(ws + 1 * halfBuf);
    _Float16* VT  = (_Float16*)(ws + 2 * halfBuf);
    _Float16* Tt  = (_Float16*)(ws + 3 * halfBuf);
    float* rowmax = (float*)(ws + 4 * halfBuf);
    float* rowsum = rowmax + (size_t)B_DIM * N_DIM;
    float* cmean  = rowsum + (size_t)B_DIM * N_DIM;
    float* cinv   = cmean  + (size_t)B_DIM * C_DIM;
    _Float16* wf_f = (_Float16*)(cinv + (size_t)B_DIM * C_DIM);
    _Float16* wf_g = wf_f + wElems;
    _Float16* wf_h = wf_g + wElems;
    _Float16* wf_o = wf_h + wElems;

    const int tiles = B_DIM * (N_DIM / 16);  // 512 blocks of 8 waves
    const int wblk  = (int)((wElems + 255) / 256);
    dim3 blk(256);

    wcvt_kernel<<<wblk, blk, 0, stream>>>(f_w, wf_f, (int)wElems);
    wcvt_kernel<<<wblk, blk, 0, stream>>>(g_w, wf_g, (int)wElems);
    wcvt_kernel<<<wblk, blk, 0, stream>>>(h_w, wf_h, (int)wElems);
    wcvt_kernel<<<wblk, blk, 0, stream>>>(o_w, wf_o, (int)wElems);

    proj_kernel<<<tiles, blk, 0, stream>>>(content, wf_f, f_b, Q, 0);
    proj_kernel<<<tiles, blk, 0, stream>>>(style,   wf_g, g_b, Kb, 0);
    proj_kernel<<<tiles, blk, 0, stream>>>(style,   wf_h, h_b, VT, 1);
    stats_kernel<<<B_DIM * C_DIM, blk, 0, stream>>>(content, cmean, cinv);
    smstats_kernel<<<tiles, blk, 0, stream>>>(Q, Kb, rowmax, rowsum);
    pv_kernel<<<tiles, blk, 0, stream>>>(Q, Kb, VT, rowmax, rowsum,
                                         content, cmean, cinv, Tt);
    outconv_kernel<<<tiles, blk, 0, stream>>>(Tt, o_w ? wf_o : wf_o, o_b, content, out);
}
